// ChildSumTreeLSTM_5016521801958
// MI455X (gfx1250) — matrix operand order repaired and checked by reading, not compiled, not run on hardware
//
#include <hip/hip_runtime.h>
#include <hip/hip_bf16.h>
#include <math.h>

typedef __attribute__((ext_vector_type(16))) __bf16 v16bf;
typedef __attribute__((ext_vector_type(8)))  __bf16 v8bf;
typedef __attribute__((ext_vector_type(8)))  float  v8f;

#define BATCH 256
#define NNODE 511
#define DIN   300
#define KX    320   // DIN zero-padded to multiple of 32
#define HDIM  512
#define KH    512
#define KTOT  832   // KX + KH
#define LDS_STRIDE 40   // row stride (bf16 elems): 32 data + 8 pad, keeps 16B alignment

__device__ __forceinline__ float sigmf(float x){ return 1.f / (1.f + __expf(-x)); }
__device__ __forceinline__ float tanhff(float x){
  float xc = fminf(fmaxf(x, -15.f), 15.f);
  float t = __expf(2.f * xc);
  return (t - 1.f) / (t + 1.f);
}
__device__ __forceinline__ v8f vzero(){
  v8f z;
  #pragma unroll
  for (int i = 0; i < 8; ++i) z[i] = 0.f;
  return z;
}

// ---- CDNA5 async global->LDS copy (16B per lane, tracked by ASYNCcnt) ----
// LDS dest: 32-bit LDS offset in VGPR (low dword of generic pointer to __shared__);
// global: GVS mode = uniform 64-bit SGPR base + per-lane 32-bit byte offset.
__device__ __forceinline__ void async_b128(const __bf16* lds_dst, const void* gbase,
                                           unsigned byte_off){
  const unsigned l = (unsigned)(size_t)lds_dst;
  asm volatile("global_load_async_to_lds_b128 %0, %1, %2"
               :: "v"(l), "v"(byte_off), "s"((unsigned long long)(size_t)gbase)
               : "memory");
}
// producer drains its async copies, then the workgroup barrier publishes LDS
__device__ __forceinline__ void wait_async_and_sync(){
  asm volatile("s_wait_asynccnt 0x0" ::: "memory");
  __syncthreads();
}

// A-matrix 16x32 bf16 fragment (CDNA5 layout):
// lane half 0: K 0..7 (elems 0..7), K 16..23 (elems 8..15); half 1: +8
__device__ __forceinline__ v16bf ld_fragA(const __bf16* row, int khalf){
  v8bf lo = *(const v8bf*)(row + khalf * 8);
  v8bf hi = *(const v8bf*)(row + 16 + khalf * 8);
  v16bf r;
  #pragma unroll
  for (int i = 0; i < 8; ++i){ r[i] = lo[i]; r[8 + i] = hi[i]; }
  return r;
}
// B-matrix 32x16 bf16 fragment: lane half 0: K 0..7, 16..23; half 1: +8
__device__ __forceinline__ v16bf ld_fragB(const __bf16* row, int khalf){
  v8bf lo = *(const v8bf*)(row + khalf * 8);
  v8bf hi = *(const v8bf*)(row + 16 + khalf * 8);
  v16bf r;
  #pragma unroll
  for (int i = 0; i < 8; ++i){ r[i] = lo[i]; r[8 + i] = hi[i]; }
  return r;
}

// ---- LDS staging ----
// pure copy of one row's 32-elem K chunk: async, bypasses VGPRs
__device__ __forceinline__ void stage_row_async(const __bf16* sA, const __bf16* __restrict__ src,
                                                int row_stride, int row){
  const unsigned goff = (unsigned)((size_t)row * row_stride * 2);
  const __bf16* d = sA + row * LDS_STRIDE;
  async_b128(d,      src, goff);
  async_b128(d + 8,  src, goff + 16);
  async_b128(d + 16, src, goff + 32);
  async_b128(d + 24, src, goff + 48);
}
// child-sum staging must pass through VGPRs (v_pk_add_bf16)
__device__ __forceinline__ void stage_row_sum(__bf16* sA, const __bf16* __restrict__ s0p,
                                              const __bf16* __restrict__ s1p,
                                              int row_stride, int row){
  const __bf16* a = s0p + (size_t)row * row_stride;
  const __bf16* b = s1p + (size_t)row * row_stride;
  __bf16* d = sA + row * LDS_STRIDE;
  #pragma unroll
  for (int j = 0; j < 32; j += 8){
    v8bf x0 = *(const v8bf*)(a + j);
    v8bf x1 = *(const v8bf*)(b + j);
    v8bf o;
    #pragma unroll
    for (int i = 0; i < 8; ++i) o[i] = (__bf16)((float)x0[i] + (float)x1[i]);
    *(v8bf*)(d + j) = o;
  }
}
__device__ __forceinline__ void stage_B_async(const __bf16* sB, const __bf16* __restrict__ Wt,
                                              int ncols, int col_base, int kg, int tid){
  if (tid < ncols * 2){
    const int col = tid >> 1;
    const int k0  = (tid & 1) * 16;
    const unsigned goff = (unsigned)(((size_t)(col_base + col) * KTOT + kg + k0) * 2);
    const __bf16* dst = sB + col * LDS_STRIDE + k0;
    async_b128(dst,     Wt, goff);
    async_b128(dst + 8, Wt, goff + 16);
  }
}

// ---- one-time repacking kernels ----
// x: (B,N,300) f32 -> xp: [node][b][320] bf16, zero-padded K
__global__ void __launch_bounds__(256) pack_x_kernel(const float* __restrict__ x,
                                                     __bf16* __restrict__ xp){
  const size_t t = (size_t)blockIdx.x * 256 + threadIdx.x;
  if (t >= (size_t)NNODE * BATCH * KX) return;
  const int k = (int)(t % KX);
  const size_t rb = t / KX;
  const int b = (int)(rb % BATCH);
  const int node = (int)(rb / BATCH);
  const float v = (k < DIN) ? x[((size_t)b * NNODE + node) * DIN + k] : 0.f;
  xp[t] = (__bf16)v;
}
// Wc: [1536 cols][832 k] bf16, cols gate-block-permuted (16 h of i|o|u interleaved)
__global__ void __launch_bounds__(256) build_wc_kernel(const float* __restrict__ W_ioux,
                                                       const float* __restrict__ W_iouh,
                                                       __bf16* __restrict__ Wc){
  const int t = blockIdx.x * 256 + threadIdx.x;
  if (t >= 3 * HDIM * KTOT) return;
  const int c = t / KTOT, k = t % KTOT;
  const int hblk = c / 48, rem = c % 48, g = rem / 16, h16 = rem % 16;
  const int outcol = g * HDIM + hblk * 16 + h16;
  float v = 0.f;
  if (k < DIN)      v = W_ioux[(size_t)k * (3 * HDIM) + outcol];
  else if (k >= KX) v = W_iouh[(size_t)(k - KX) * (3 * HDIM) + outcol];
  Wc[t] = (__bf16)v;
}
__global__ void __launch_bounds__(256) build_wf_kernel(const float* __restrict__ W_fx,
                                                       const float* __restrict__ W_fh,
                                                       __bf16* __restrict__ Wf){
  const int t = blockIdx.x * 256 + threadIdx.x;
  if (t >= HDIM * KTOT) return;
  const int c = t / KTOT, k = t % KTOT;
  float v = 0.f;
  if (k < DIN)      v = W_fx[(size_t)k * HDIM + c];
  else if (k >= KX) v = W_fh[(size_t)(k - KX) * HDIM + c];
  Wf[t] = (__bf16)v;
}

// ---- iou GEMM + fused LSTM cell epilogue ----
// grid: (nL, 32); block 256 (8 waves). Tile: 256 rows (one node's batch) x 48 cols.
// Each wave: 2 m-subtiles x 3 gate-subtiles = 6 WMMAs per 32-K chunk. LDS double-buffered,
// pure-copy stages use async global->LDS.
__global__ void __launch_bounds__(256) iou_gemm_kernel(
    const __bf16* __restrict__ xp, const __bf16* __restrict__ Wc,
    const float* __restrict__ b_ioux, const float* __restrict__ b_iouh,
    const __bf16* __restrict__ h_prev, const float* __restrict__ fc_sum,
    float* __restrict__ c_cur, __bf16* __restrict__ h_cur,
    float* __restrict__ out, int base, int has_children, int is_root)
{
  __shared__ __align__(16) __bf16 sA[2][256 * LDS_STRIDE];
  __shared__ __align__(16) __bf16 sB[2][48 * LDS_STRIDE];
  const int tid = threadIdx.x;
  const int wave = tid >> 5, lane = tid & 31;
  const int khalf = lane >> 4, nIdx = lane & 15;
  const int p = blockIdx.x, bn = blockIdx.y;
  const int r0 = p * 256;
  const int Kt = has_children ? KTOT : KX;

  const __bf16* xrow = xp + (size_t)(base + p) * BATCH * KX;          // [256][KX]
  const __bf16* h0row = h_prev + (size_t)(2 * p) * BATCH * HDIM;      // child0 [256][HDIM]
  const __bf16* h1row = h0row + (size_t)BATCH * HDIM;                 // child1

  v8f acc[2][3];
  #pragma unroll
  for (int m = 0; m < 2; ++m)
    #pragma unroll
    for (int s = 0; s < 3; ++s) acc[m][s] = vzero();

  // prologue: stage chunk 0 into buffer 0
  stage_row_async(sA[0], xrow, KX, tid);
  stage_B_async(sB[0], Wc, 48, bn * 48, 0, tid);

  for (int kg = 0; kg < Kt; kg += 32){
    const int buf = (kg >> 5) & 1;
    wait_async_and_sync();
    const int kn = kg + 32;
    if (kn < Kt){                                  // stage next chunk into other buffer
      if (kn < KX) stage_row_async(sA[buf ^ 1], xrow + kn, KX, tid);
      else         stage_row_sum(sA[buf ^ 1], h0row + (kn - KX), h1row + (kn - KX), HDIM, tid);
      stage_B_async(sB[buf ^ 1], Wc, 48, bn * 48, kn, tid);
    }
    const __bf16* sa = sA[buf];
    const __bf16* sb = sB[buf];
    const v16bf a0 = ld_fragA(sa + (wave * 32 + nIdx) * LDS_STRIDE, khalf);
    const v16bf a1 = ld_fragA(sa + (wave * 32 + 16 + nIdx) * LDS_STRIDE, khalf);
    #pragma unroll
    for (int s = 0; s < 3; ++s){
      const v16bf bb = ld_fragB(sb + (s * 16 + nIdx) * LDS_STRIDE, khalf);
      acc[0][s] = __builtin_amdgcn_wmma_f32_16x16x32_bf16(false, a0, false, bb,
                                                          (short)0, acc[0][s], false, false);
      acc[1][s] = __builtin_amdgcn_wmma_f32_16x16x32_bf16(false, a1, false, bb,
                                                          (short)0, acc[1][s], false, false);
    }
  }

  // epilogue: acc[.][0]=i, acc[.][1]=o, acc[.][2]=u for h = bn*16 + nIdx
  const int h = bn * 16 + nIdx;
  const float bi = b_ioux[h]            + b_iouh[h];
  const float bo = b_ioux[HDIM + h]     + b_iouh[HDIM + h];
  const float bu = b_ioux[2 * HDIM + h] + b_iouh[2 * HDIM + h];
  #pragma unroll
  for (int m = 0; m < 2; ++m){
    #pragma unroll
    for (int e = 0; e < 8; ++e){
      const int r = r0 + wave * 32 + m * 16 + khalf * 8 + e;
      const size_t o = (size_t)r * HDIM + h;
      const float fc = has_children ? fc_sum[o] : 0.f;
      const float cn = sigmf(acc[m][0][e] + bi) * tanhff(acc[m][2][e] + bu) + fc;
      const float hn = sigmf(acc[m][1][e] + bo) * tanhff(cn);
      c_cur[o] = cn;
      h_cur[o] = (__bf16)hn;
      if (is_root){
        const size_t ob = (size_t)(r & 255) * HDIM + h;
        out[ob] = cn;                              // c[:,0]
        out[(size_t)BATCH * HDIM + ob] = hn;       // h[:,0]
      }
    }
  }
}

// ---- forget-gate GEMM: fc_sum = sum_j sigmoid([x;h_j] @ Wf + b) * c_j ----
// grid: (M/128, 16); tile 128 x 32, LDS double-buffered, x@W_fx shared across children.
__global__ void __launch_bounds__(256) fgate_kernel(
    const __bf16* __restrict__ xp, const __bf16* __restrict__ Wf,
    const float* __restrict__ b_fx, const float* __restrict__ b_fh,
    const __bf16* __restrict__ h_prev, const float* __restrict__ c_prev,
    float* __restrict__ fc_sum, int base)
{
  __shared__ __align__(16) __bf16 sA[2][128 * LDS_STRIDE];
  __shared__ __align__(16) __bf16 sB[2][32 * LDS_STRIDE];
  const int tid = threadIdx.x;
  const int wave = tid >> 5, lane = tid & 31;
  const int khalf = lane >> 4, nIdx = lane & 15;
  const int r0 = blockIdx.x * 128, bn = blockIdx.y;
  const int p = r0 >> 8, b0 = r0 & 255;

  const __bf16* xrow  = xp + ((size_t)(base + p) * BATCH + b0) * KX;
  const __bf16* h0row = h_prev + ((size_t)(2 * p) * BATCH + b0) * HDIM;
  const __bf16* h1row = h0row + (size_t)BATCH * HDIM;

  v8f accx[2] = { vzero(), vzero() };
  v8f acc0[2], acc1[2];

  // ---- phase X: shared x @ W_fx partial (K = 320) ----
  if (tid < 128) stage_row_async(sA[0], xrow, KX, tid);
  stage_B_async(sB[0], Wf, 32, bn * 32, 0, tid);
  for (int kg = 0; kg < KX; kg += 32){
    const int buf = (kg >> 5) & 1;
    wait_async_and_sync();
    if (kg + 32 < KX){
      if (tid < 128) stage_row_async(sA[buf ^ 1], xrow + kg + 32, KX, tid);
      stage_B_async(sB[buf ^ 1], Wf, 32, bn * 32, kg + 32, tid);
    }
    const v16bf a = ld_fragA(sA[buf] + (wave * 16 + nIdx) * LDS_STRIDE, khalf);
    #pragma unroll
    for (int s = 0; s < 2; ++s){
      const v16bf bb = ld_fragB(sB[buf] + (s * 16 + nIdx) * LDS_STRIDE, khalf);
      accx[s] = __builtin_amdgcn_wmma_f32_16x16x32_bf16(false, a, false, bb,
                                                        (short)0, accx[s], false, false);
    }
  }
  #pragma unroll
  for (int s = 0; s < 2; ++s) acc0[s] = accx[s];

  // ---- phase child 0: h0 @ W_fh (K = 512) ----
  wait_async_and_sync();
  if (tid < 128) stage_row_async(sA[0], h0row, HDIM, tid);
  stage_B_async(sB[0], Wf, 32, bn * 32, KX, tid);
  for (int kg = 0; kg < KH; kg += 32){
    const int buf = (kg >> 5) & 1;
    wait_async_and_sync();
    if (kg + 32 < KH){
      if (tid < 128) stage_row_async(sA[buf ^ 1], h0row + kg + 32, HDIM, tid);
      stage_B_async(sB[buf ^ 1], Wf, 32, bn * 32, KX + kg + 32, tid);
    }
    const v16bf a = ld_fragA(sA[buf] + (wave * 16 + nIdx) * LDS_STRIDE, khalf);
    #pragma unroll
    for (int s = 0; s < 2; ++s){
      const v16bf bb = ld_fragB(sB[buf] + (s * 16 + nIdx) * LDS_STRIDE, khalf);
      acc0[s] = __builtin_amdgcn_wmma_f32_16x16x32_bf16(false, a, false, bb,
                                                        (short)0, acc0[s], false, false);
    }
  }
  #pragma unroll
  for (int s = 0; s < 2; ++s) acc1[s] = accx[s];

  // ---- phase child 1 ----
  wait_async_and_sync();
  if (tid < 128) stage_row_async(sA[0], h1row, HDIM, tid);
  stage_B_async(sB[0], Wf, 32, bn * 32, KX, tid);
  for (int kg = 0; kg < KH; kg += 32){
    const int buf = (kg >> 5) & 1;
    wait_async_and_sync();
    if (kg + 32 < KH){
      if (tid < 128) stage_row_async(sA[buf ^ 1], h1row + kg + 32, HDIM, tid);
      stage_B_async(sB[buf ^ 1], Wf, 32, bn * 32, KX + kg + 32, tid);
    }
    const v16bf a = ld_fragA(sA[buf] + (wave * 16 + nIdx) * LDS_STRIDE, khalf);
    #pragma unroll
    for (int s = 0; s < 2; ++s){
      const v16bf bb = ld_fragB(sB[buf] + (s * 16 + nIdx) * LDS_STRIDE, khalf);
      acc1[s] = __builtin_amdgcn_wmma_f32_16x16x32_bf16(false, a, false, bb,
                                                        (short)0, acc1[s], false, false);
    }
  }

  // ---- epilogue: fc_sum = sigma(f0)*c0 + sigma(f1)*c1 ----
  #pragma unroll
  for (int s = 0; s < 2; ++s){
    const int h = bn * 32 + s * 16 + nIdx;
    const float bias = b_fx[h] + b_fh[h];
    #pragma unroll
    for (int e = 0; e < 8; ++e){
      const int r = r0 + wave * 16 + khalf * 8 + e;
      const int b = r & 255;
      const size_t rc0 = ((size_t)(2 * p) * BATCH + b) * HDIM + h;
      const size_t rc1 = rc0 + (size_t)BATCH * HDIM;
      const float f0 = sigmf(acc0[s][e] + bias);
      const float f1 = sigmf(acc1[s][e] + bias);
      fc_sum[(size_t)r * HDIM + h] = f0 * c_prev[rc0] + f1 * c_prev[rc1];
    }
  }
}

extern "C" void kernel_launch(void* const* d_in, const int* in_sizes, int n_in,
                              void* d_out, int out_size, void* d_ws, size_t ws_size,
                              hipStream_t stream){
  (void)in_sizes; (void)n_in; (void)out_size; (void)ws_size;
  const float* x      = (const float*)d_in[0];
  const float* W_ioux = (const float*)d_in[1];
  const float* b_ioux = (const float*)d_in[2];
  const float* W_iouh = (const float*)d_in[3];
  const float* b_iouh = (const float*)d_in[4];
  const float* W_fx   = (const float*)d_in[5];
  const float* b_fx   = (const float*)d_in[6];
  const float* W_fh   = (const float*)d_in[7];
  const float* b_fh   = (const float*)d_in[8];
  float* out = (float*)d_out;

  // workspace carve-out (~560 MB): packed bf16 x + weights, ping-pong h(bf16)/c(f32)
  // level buffers (max 256 nodes/level), per-level fc_sum buffer.
  char* ws = (char*)d_ws;
  size_t off = 0;
  auto carve = [&](size_t bytes) -> char* {
    char* pcur = ws + off;
    off = (off + bytes + 255) & ~(size_t)255;
    return pcur;
  };
  __bf16* xp = (__bf16*)carve((size_t)NNODE * BATCH * KX * 2);
  __bf16* Wc = (__bf16*)carve((size_t)3 * HDIM * KTOT * 2);
  __bf16* Wf = (__bf16*)carve((size_t)HDIM * KTOT * 2);
  __bf16* h_buf[2];
  float*  c_buf[2];
  h_buf[0] = (__bf16*)carve((size_t)BATCH * 256 * HDIM * 2);
  h_buf[1] = (__bf16*)carve((size_t)BATCH * 256 * HDIM * 2);
  c_buf[0] = (float*)carve((size_t)BATCH * 256 * HDIM * 4);
  c_buf[1] = (float*)carve((size_t)BATCH * 256 * HDIM * 4);
  float* fc = (float*)carve((size_t)BATCH * 128 * HDIM * 4);

  {
    const size_t total = (size_t)NNODE * BATCH * KX;
    pack_x_kernel<<<(unsigned)((total + 255) / 256), 256, 0, stream>>>(x, xp);
  }
  build_wc_kernel<<<(3 * HDIM * KTOT + 255) / 256, 256, 0, stream>>>(W_ioux, W_iouh, Wc);
  build_wf_kernel<<<(HDIM * KTOT + 255) / 256, 256, 0, stream>>>(W_fx, W_fh, Wf);

  for (int level = 8; level >= 0; --level){
    const int nL = 1 << level;
    const int M = BATCH * nL;
    const int base = nL - 1;
    const int cur = level & 1, prv = cur ^ 1;
    if (level < 8)
      fgate_kernel<<<dim3(M / 128, HDIM / 32), 256, 0, stream>>>(
          xp, Wf, b_fx, b_fh, h_buf[prv], c_buf[prv], fc, base);
    iou_gemm_kernel<<<dim3(nL, (3 * HDIM) / 48), 256, 0, stream>>>(
        xp, Wc, b_ioux, b_iouh, h_buf[prv], fc, c_buf[cur], h_buf[cur],
        out, base, level < 8 ? 1 : 0, level == 0 ? 1 : 0);
  }
}
